// MultiHeadedAttention_88708254532001
// MI455X (gfx1250) — matrix operand order repaired and checked by reading, not compiled, not run on hardware
//
#include <hip/hip_runtime.h>
#include <hip/hip_bf16.h>

// ---------------- types ----------------
typedef __attribute__((ext_vector_type(16))) __bf16 v16bf;
typedef __attribute__((ext_vector_type(8)))  __bf16 v8bf;
typedef __attribute__((ext_vector_type(8)))  float  v8f;

union ABf { v16bf v; v8bf h[2]; };

#define BB  8
#define SS  1024
#define UU  1024
#define HH  16
#define DKK 64

__device__ inline v8f wmma_bf16(const ABf& a, const ABf& b, v8f c) {
    // D = A(16x32) * B(32x16) + C(16x16), f32 accumulate
    return __builtin_amdgcn_wmma_f32_16x16x32_bf16(
        false, a.v, false, b.v, (short)0, c, false, false);
}

// ---------------- conversion kernels ----------------
__global__ void cvt_f32_bf16(const float* __restrict__ in, __bf16* __restrict__ out, int n) {
    int i = blockIdx.x * blockDim.x + threadIdx.x;
    if (i < n) out[i] = (__bf16)in[i];
}

// in: [batch, R, C] f32 -> out: [batch, C, R] bf16 (per-batch transpose)
__global__ void transpose_f32_bf16(const float* __restrict__ in, __bf16* __restrict__ out,
                                   int batch, int R, int C) {
    int i = blockIdx.x * blockDim.x + threadIdx.x;
    int total = batch * R * C;
    if (i >= total) return;
    int bh = i / (R * C);
    int rc = i % (R * C);
    int r = rc / C, c = rc % C;
    out[(size_t)bh * R * C + (size_t)c * R + r] = (__bf16)in[i];
}

// ---------------- GEMM: out = A(MxK) * Bt(NxK)^T + bias ----------------
// Per wave: 64 rows x 64 cols (4 A strips x 4 B tiles = 16 WMMA / k-step).
// Block: 4 waves -> 256 rows x 64 cols macro tile.
// MODE 0: bf16 out row-major (MxN)
// MODE 1: bf16 out transposed per batch of SS rows: out[b, col, s]
// MODE 2: f32 out row-major
template<int MODE>
__global__ __launch_bounds__(128)
void gemm_bf16(const __bf16* __restrict__ A, const __bf16* __restrict__ Bt,
               const float* __restrict__ bias, void* __restrict__ Out,
               int M, int N, int K)
{
    const int lane = threadIdx.x & 31;
    const int wv   = threadIdx.x >> 5;
    const int g    = lane >> 4;      // half-wave group
    const int lm   = lane & 15;
    const int row0 = blockIdx.x * 256 + wv * 64;
    const int col0 = blockIdx.y * 64;

    v8f c[4][4];
#pragma unroll
    for (int ar = 0; ar < 4; ++ar)
#pragma unroll
        for (int bt = 0; bt < 4; ++bt)
            c[ar][bt] = v8f{0.f,0.f,0.f,0.f,0.f,0.f,0.f,0.f};

    const __bf16* arow0 = A + (size_t)(row0 + lm) * K;
    for (int k0 = 0; k0 < K; k0 += 32) {
        // A operands: strip ar covers rows row0 + ar*16 .. +15
        ABf a[4];
#pragma unroll
        for (int ar = 0; ar < 4; ++ar) {
            const __bf16* ap = arow0 + (size_t)(ar * 16) * K + k0 + g * 8;
            a[ar].h[0] = *(const v8bf*)(ap);
            a[ar].h[1] = *(const v8bf*)(ap + 16);
        }
        // B operands: tile bt covers cols col0 + bt*16 .. +15
        ABf b[4];
#pragma unroll
        for (int bt = 0; bt < 4; ++bt) {
            const __bf16* bp = Bt + (size_t)(col0 + bt * 16 + lm) * K + k0 + g * 8;
            b[bt].h[0] = *(const v8bf*)(bp);
            b[bt].h[1] = *(const v8bf*)(bp + 16);
        }
#pragma unroll
        for (int ar = 0; ar < 4; ++ar)
#pragma unroll
            for (int bt = 0; bt < 4; ++bt)
                c[ar][bt] = wmma_bf16(a[ar], b[bt], c[ar][bt]);
    }

#pragma unroll
    for (int ar = 0; ar < 4; ++ar) {
#pragma unroll
        for (int bt = 0; bt < 4; ++bt) {
#pragma unroll
            for (int r = 0; r < 8; ++r) {
                int row = row0 + ar * 16 + r + g * 8;
                int col = col0 + bt * 16 + lm;
                float val = c[ar][bt][r] + bias[col];
                if (MODE == 0) {
                    ((__bf16*)Out)[(size_t)row * N + col] = (__bf16)val;
                } else if (MODE == 1) {
                    int b2 = row >> 10;          // row / SS
                    int s  = row & (SS - 1);     // row % SS
                    ((__bf16*)Out)[(size_t)b2 * N * SS + (size_t)col * SS + s] = (__bf16)val;
                } else {
                    ((float*)Out)[(size_t)row * N + col] = val;
                }
            }
        }
    }
}

// ---------------- flash attention (causal) ----------------
// Q, Kb: [B*S, U] bf16 (head h at cols h*64..h*64+63)
// Vt:    [B, U, S] bf16 (transposed V)
// O:     [B*S, U] bf16
__global__ __launch_bounds__(128)
void attn_fwd(const __bf16* __restrict__ Q, const __bf16* __restrict__ Kb,
              const __bf16* __restrict__ Vt, __bf16* __restrict__ O)
{
    __shared__ __align__(16) __bf16 pbuf[4][16][32];
    const int lane = threadIdx.x & 31;
    const int wv   = threadIdx.x >> 5;
    const int g    = lane >> 4;
    const int lm   = lane & 15;

    const int QT = SS / 16;  // q-tiles per (b,h)
    int tile = __builtin_amdgcn_readfirstlane((int)blockIdx.x * 4 + wv);
    int qt = tile % QT;
    int h  = (tile / QT) % HH;
    int b  = tile / (QT * HH);
    int qbase = qt * 16;

    // Q A-operands (16 q-rows x 64 d, split into two K=32 operands)
    const __bf16* qrow = Q + ((size_t)b * SS + qbase + lm) * UU + h * DKK;
    ABf aq[2];
#pragma unroll
    for (int t = 0; t < 2; ++t) {
        aq[t].h[0] = *(const v8bf*)(qrow + t * 32 + g * 8);
        aq[t].h[1] = *(const v8bf*)(qrow + t * 32 + g * 8 + 16);
    }

    float m[8], l[8];
    v8f o[4];
#pragma unroll
    for (int r = 0; r < 8; ++r) { m[r] = -1e30f; l[r] = 0.f; }
#pragma unroll
    for (int t = 0; t < 4; ++t) o[t] = v8f{0.f,0.f,0.f,0.f,0.f,0.f,0.f,0.f};

    const float scale = 0.125f;  // 1/sqrt(64)
    const __bf16* kbase = Kb + (size_t)b * SS * UU + h * DKK;

    for (int j0 = 0; j0 <= qbase + 15; j0 += 32) {
        // scores: two 16x16 tiles (k cols j0.., j0+16..)
        v8f s0 = v8f{0.f,0.f,0.f,0.f,0.f,0.f,0.f,0.f};
        v8f s1 = v8f{0.f,0.f,0.f,0.f,0.f,0.f,0.f,0.f};
#pragma unroll
        for (int t = 0; t < 2; ++t) {
            // B operand = K^T tile; lane column k = j0(+16)+lm, d ranges t*32 + g*8 and t*32 + g*8 + 16
            const __bf16* p0 = kbase + (size_t)(j0 + lm) * UU + t * 32 + g * 8;
            const __bf16* p1 = kbase + (size_t)(j0 + 16 + lm) * UU + t * 32 + g * 8;
            ABf bk0, bk1;
            bk0.h[0] = *(const v8bf*)p0; bk0.h[1] = *(const v8bf*)(p0 + 16);
            bk1.h[0] = *(const v8bf*)p1; bk1.h[1] = *(const v8bf*)(p1 + 16);
            s0 = wmma_bf16(aq[t], bk0, s0);
            s1 = wmma_bf16(aq[t], bk1, s1);
        }
        // online softmax per row r (row = qbase + r + 8*g)
#pragma unroll
        for (int r = 0; r < 8; ++r) {
            int q = qbase + r + g * 8;
            float x0 = (j0 + lm      <= q) ? s0[r] * scale : -1e9f;
            float x1 = (j0 + 16 + lm <= q) ? s1[r] * scale : -1e9f;
            float t = fmaxf(x0, x1);
            t = fmaxf(t, __shfl_xor(t, 1));
            t = fmaxf(t, __shfl_xor(t, 2));
            t = fmaxf(t, __shfl_xor(t, 4));
            t = fmaxf(t, __shfl_xor(t, 8));
            float mnew = fmaxf(m[r], t);
            float corr = __expf(m[r] - mnew);
            m[r] = mnew;
            float e0 = __expf(x0 - mnew);
            float e1 = __expf(x1 - mnew);
            float ls = e0 + e1;
            ls += __shfl_xor(ls, 1);
            ls += __shfl_xor(ls, 2);
            ls += __shfl_xor(ls, 4);
            ls += __shfl_xor(ls, 8);
            l[r] = l[r] * corr + ls;
#pragma unroll
            for (int t2 = 0; t2 < 4; ++t2) o[t2][r] *= corr;
            // stage P (C-layout position) into LDS for C->A transpose
            pbuf[wv][r + g * 8][lm]      = (__bf16)e0;
            pbuf[wv][r + g * 8][16 + lm] = (__bf16)e1;
        }
        // read P back in A-operand layout (same wave; DS ops in-order)
        ABf ap;
        ap.h[0] = *(const v8bf*)&pbuf[wv][lm][g * 8];
        ap.h[1] = *(const v8bf*)&pbuf[wv][lm][g * 8 + 16];
        // O += P (16x32) * V (32x64)
#pragma unroll
        for (int t = 0; t < 4; ++t) {
            const __bf16* vp = Vt + ((size_t)b * UU + h * DKK + t * 16 + lm) * SS + j0 + g * 8;
            ABf bv;
            bv.h[0] = *(const v8bf*)vp; bv.h[1] = *(const v8bf*)(vp + 16);
            o[t] = wmma_bf16(ap, bv, o[t]);
        }
    }
    // epilogue: normalize and store (B,S,U) with head h at cols h*64..
#pragma unroll
    for (int r = 0; r < 8; ++r) {
        float inv = 1.f / l[r];
#pragma unroll
        for (int t = 0; t < 4; ++t) {
            O[((size_t)b * SS + qbase + r + g * 8) * UU + h * DKK + t * 16 + lm] =
                (__bf16)(o[t][r] * inv);
        }
    }
}

// ---------------- launcher ----------------
extern "C" void kernel_launch(void* const* d_in, const int* in_sizes, int n_in,
                              void* d_out, int out_size, void* d_ws, size_t ws_size,
                              hipStream_t stream) {
    (void)in_sizes; (void)n_in; (void)out_size; (void)ws_size;
    const float* query = (const float*)d_in[0];
    // d_in[1]=key, d_in[2]=value, d_in[3]=mask: unused (reference projects `query` for q,k,v; mask is causal)
    const float* Wq = (const float*)d_in[4];
    const float* bq = (const float*)d_in[5];
    const float* Wk = (const float*)d_in[6];
    const float* bk = (const float*)d_in[7];
    const float* Wv = (const float*)d_in[8];
    const float* bv = (const float*)d_in[9];
    const float* Wo = (const float*)d_in[10];
    const float* bo = (const float*)d_in[11];

    const size_t NX = (size_t)BB * SS * UU;   // 8M elements
    const size_t NW = (size_t)UU * UU;        // 1M elements
    char* ws = (char*)d_ws;
    __bf16* Xbf = (__bf16*)ws;                // 16 MB
    __bf16* Wqt = Xbf + NX;                   // 2 MB each
    __bf16* Wkt = Wqt + NW;
    __bf16* Wvt = Wkt + NW;
    __bf16* Wot = Wvt + NW;
    __bf16* Qb  = Wot + NW;                   // 16 MB
    __bf16* Kb  = Qb + NX;                    // 16 MB
    __bf16* Vt  = Kb + NX;                    // 16 MB
    __bf16* Ao  = Vt + NX;                    // 16 MB

    // convert / transpose to bf16
    cvt_f32_bf16<<<(int)((NX + 255) / 256), 256, 0, stream>>>(query, Xbf, (int)NX);
    transpose_f32_bf16<<<(int)((HH * UU * DKK + 255) / 256), 256, 0, stream>>>(Wq, Wqt, HH, UU, DKK);
    transpose_f32_bf16<<<(int)((HH * UU * DKK + 255) / 256), 256, 0, stream>>>(Wk, Wkt, HH, UU, DKK);
    transpose_f32_bf16<<<(int)((HH * UU * DKK + 255) / 256), 256, 0, stream>>>(Wv, Wvt, HH, UU, DKK);
    transpose_f32_bf16<<<(int)((UU * UU + 255) / 256), 256, 0, stream>>>(Wo, Wot, 1, UU, UU);

    const int M = BB * SS;  // 8192
    dim3 gg(M / 256, UU / 64);  // (32, 16)
    gemm_bf16<0><<<gg, 128, 0, stream>>>(Xbf, Wqt, bq, (void*)Qb, M, UU, UU);
    gemm_bf16<0><<<gg, 128, 0, stream>>>(Xbf, Wkt, bk, (void*)Kb, M, UU, UU);
    gemm_bf16<1><<<gg, 128, 0, stream>>>(Xbf, Wvt, bv, (void*)Vt, M, UU, UU);

    attn_fwd<<<BB * HH * (SS / 16) / 4, 128, 0, stream>>>(Qb, Kb, Vt, Ao);

    gemm_bf16<2><<<gg, 128, 0, stream>>>(Ao, Wot, bo, d_out, M, UU, UU);
}
